// QLSTM_86122684219686
// MI455X (gfx1250) — matrix operand order repaired and checked by reading, not compiled, not run on hardware
//
#include <hip/hip_runtime.h>
#include <hip/hip_bf16.h>

typedef __attribute__((ext_vector_type(16))) __bf16 v16bf;
typedef __attribute__((ext_vector_type(8)))  __bf16 v8bf;
typedef __attribute__((ext_vector_type(8)))  float  v8f;

#define T_STEPS 100
#define BATCH   128
#define FEAT    256
#define HID     1024
#define NKT     40          // (FEAT+HID)/32 K-tiles
#define NNT     64          // HID/16 N-tiles per gate

static __device__ __forceinline__ __bf16 f2bf(float f) {
    unsigned u = __builtin_bit_cast(unsigned, f);
    u += 0x7FFFu + ((u >> 16) & 1u);                 // round-to-nearest-even
    unsigned short s = (unsigned short)(u >> 16);
    return __builtin_bit_cast(__bf16, s);
}

// A-operand (16-bit, 16x32) per CDNA5 ISA: lane L holds row m=L%16;
// elements 0..7 = K {8h..8h+7}, elements 8..15 = K {16+8h..16+8h+7}, h=L/16.
// With row-major bf16 storage that is two contiguous 16B chunks.
static __device__ __forceinline__ v16bf load_a(const __bf16* p) {
    v8bf lo = *(const v8bf*)(p);
    v8bf hi = *(const v8bf*)(p + 16);
    v16bf r;
#pragma unroll
    for (int i = 0; i < 8; ++i) { r[i] = lo[i]; r[i + 8] = hi[i]; }
    return r;
}

// ---------------------------------------------------------------------------
// Prep: assemble quaternion Hamilton-product weights [W;U] (1280 x 1024) per
// gate, converted to bf16, stored pre-swizzled for the WMMA B operand:
// layout [gate][nt][kt][lane][16 halfwords], lane holds column n = nt*16 +
// lane%16, K values k = kt*32 + 8*(lane/16) + (e%8) + 16*(e/8).
// ---------------------------------------------------------------------------
__global__ void qlstm_prep_w(
    const float* __restrict__ wf, const float* __restrict__ wi,
    const float* __restrict__ wo, const float* __restrict__ wc,
    const float* __restrict__ uf, const float* __restrict__ ui,
    const float* __restrict__ uo, const float* __restrict__ uc,
    __bf16* __restrict__ wsw)
{
    size_t idx = (size_t)blockIdx.x * blockDim.x + threadIdx.x;
    const size_t total = (size_t)4 * NNT * NKT * 512;
    if (idx >= total) return;

    int e    = (int)(idx & 15);
    int lane = (int)((idx >> 4) & 31);
    int kt   = (int)((idx >> 9) % NKT);
    int nt   = (int)(((idx >> 9) / NKT) % NNT);
    int gate = (int)(idx / ((size_t)NNT * NKT * 512));

    int n = nt * 16 + (lane & 15);
    int k = kt * 32 + (lane >> 4) * 8 + (e & 7) + (e >> 3) * 16;

    const float* wptr[4] = {wf, wi, wo, wc};
    const float* uptr[4] = {uf, ui, uo, uc};
    // Hamilton block table: component index and sign per (col-block, row-block)
    const int   cidx[4][4] = {{0,1,2,3},{1,0,3,2},{2,3,0,1},{3,2,1,0}};
    const float csgn[4][4] = {{1.f,-1.f,-1.f,-1.f},
                              {1.f, 1.f,-1.f, 1.f},
                              {1.f, 1.f, 1.f,-1.f},
                              {1.f,-1.f, 1.f, 1.f}};

    const int P  = HID / 4;          // 256 (out quarter)
    int cb = n / P, q = n % P;
    float v;
    if (k < FEAT) {                  // input->hidden block W (in = 256)
        const int Q = FEAT / 4;      // 64
        int rb = k / Q, p = k % Q;
        v = csgn[cb][rb] * wptr[gate][((size_t)cidx[cb][rb] * Q + p) * P + q];
    } else {                         // hidden->hidden block U (in = 1024)
        const int Q = HID / 4;       // 256
        int kk = k - FEAT;
        int rb = kk / Q, p = kk % Q;
        v = csgn[cb][rb] * uptr[gate][((size_t)cidx[cb][rb] * Q + p) * P + q];
    }
    wsw[idx] = f2bf(v);
}

__global__ void qlstm_prep_x(const float* __restrict__ x,
                             __bf16* __restrict__ xbf, int n)
{
    int i = blockIdx.x * blockDim.x + threadIdx.x;
    if (i < n) xbf[i] = f2bf(x[i]);
}

__global__ void qlstm_zero(__bf16* __restrict__ hbf, float* __restrict__ c, int n)
{
    int i = blockIdx.x * blockDim.x + threadIdx.x;
    if (i < n) { hbf[i] = f2bf(0.0f); c[i] = 0.0f; }
}

__global__ void qlstm_last(const float* __restrict__ src,
                           float* __restrict__ dst, int n)
{
    int i = blockIdx.x * blockDim.x + threadIdx.x;
    if (i < n) dst[i] = src[i];
}

// ---------------------------------------------------------------------------
// One recurrent step: preact[g] = [x_t, h_{t-1}] @ [W_g; U_g] + b_g (K = 1280)
// then fused LSTM cell update. Grid: 64 blocks (one N-tile of H each) x 4
// waves; each wave: 2 M-tiles x 4 gates -> 8 f32 accumulators, 320 WMMAs.
// ---------------------------------------------------------------------------
__global__ __launch_bounds__(128) void qlstm_step(
    const __bf16* __restrict__ xbf,   // (T*B, FEAT) row-major bf16
    const __bf16* __restrict__ wsw,   // swizzled weights
    const __bf16* __restrict__ hin,   // (B, HID) bf16, h_{t-1}
    __bf16* __restrict__ hout,        // (B, HID) bf16, h_t
    float* __restrict__ cst,          // (B, HID) f32 cell state (in-place)
    const float* __restrict__ bf_, const float* __restrict__ bi_,
    const float* __restrict__ bo_, const float* __restrict__ bc_,
    float* __restrict__ out, int t)
{
    const int lane = threadIdx.x & 31;
    const int wave = threadIdx.x >> 5;      // 0..3 -> M-tile pair
    const int nt   = blockIdx.x;            // 0..63 -> 16 cols of H
    const int half = lane >> 4;
    const int nl   = lane & 15;

    v8f acc[2][4];
    const float* bptr[4] = {bf_, bi_, bo_, bc_};
#pragma unroll
    for (int g = 0; g < 4; ++g) {
        float b = bptr[g][nt * 16 + nl];
#pragma unroll
        for (int ms = 0; ms < 2; ++ms)
#pragma unroll
            for (int j = 0; j < 8; ++j) acc[ms][g][j] = b;
    }

    const __bf16* wbase  = wsw + (size_t)nt * NKT * 512;
    const size_t gstride = (size_t)NNT * NKT * 512;

    const int r0 = (wave * 2 + 0) * 16 + nl;     // batch rows (A: m = lane%16)
    const int r1 = (wave * 2 + 1) * 16 + nl;

    // ---- K part 1: x_t contribution (kt = 0..7) ----
    const __bf16* xrow0 = xbf + (size_t)(t * BATCH + r0) * FEAT + half * 8;
    const __bf16* xrow1 = xbf + (size_t)(t * BATCH + r1) * FEAT + half * 8;
#pragma unroll 2
    for (int kt = 0; kt < 8; ++kt) {
        v16bf a0 = load_a(xrow0 + kt * 32);
        v16bf a1 = load_a(xrow1 + kt * 32);
#pragma unroll
        for (int g = 0; g < 4; ++g) {
            v16bf b = *(const v16bf*)(wbase + g * gstride + (size_t)kt * 512 + lane * 16);
            acc[0][g] = __builtin_amdgcn_wmma_f32_16x16x32_bf16(
                false, a0, false, b, (short)0, acc[0][g], false, false);
            acc[1][g] = __builtin_amdgcn_wmma_f32_16x16x32_bf16(
                false, a1, false, b, (short)0, acc[1][g], false, false);
        }
    }

    // ---- K part 2: h_{t-1} contribution (kt = 8..39) ----
    const __bf16* hrow0 = hin + (size_t)r0 * HID + half * 8;
    const __bf16* hrow1 = hin + (size_t)r1 * HID + half * 8;
#pragma unroll 2
    for (int kt2 = 0; kt2 < 32; ++kt2) {
        v16bf a0 = load_a(hrow0 + kt2 * 32);
        v16bf a1 = load_a(hrow1 + kt2 * 32);
#pragma unroll
        for (int g = 0; g < 4; ++g) {
            v16bf b = *(const v16bf*)(wbase + g * gstride + (size_t)(kt2 + 8) * 512 + lane * 16);
            acc[0][g] = __builtin_amdgcn_wmma_f32_16x16x32_bf16(
                false, a0, false, b, (short)0, acc[0][g], false, false);
            acc[1][g] = __builtin_amdgcn_wmma_f32_16x16x32_bf16(
                false, a1, false, b, (short)0, acc[1][g], false, false);
        }
    }

    // ---- fused LSTM cell epilogue ----
    // C/D layout: VGPR j -> m = 8*(lane/16) + j, n = lane%16
    const int col = nt * 16 + nl;
#pragma unroll
    for (int ms = 0; ms < 2; ++ms) {
        const int rbase = (wave * 2 + ms) * 16 + half * 8;
#pragma unroll
        for (int j = 0; j < 8; ++j) {
            size_t idx = (size_t)(rbase + j) * HID + col;
            float fg = 1.0f / (1.0f + __expf(-acc[ms][0][j]));
            float ig = 1.0f / (1.0f + __expf(-acc[ms][1][j]));
            float og = 1.0f / (1.0f + __expf(-acc[ms][2][j]));
            float at = acc[ms][3][j];
            float cn = ig * tanhf(at) + fg * cst[idx];
            float hn = og * tanhf(cn);
            cst[idx] = cn;
            out[(size_t)t * BATCH * HID + idx] = hn;
            hout[idx] = f2bf(hn);
        }
    }
}

extern "C" void kernel_launch(void* const* d_in, const int* in_sizes, int n_in,
                              void* d_out, int out_size, void* d_ws, size_t ws_size,
                              hipStream_t stream)
{
    (void)in_sizes; (void)n_in; (void)out_size; (void)ws_size;
    const float* x = (const float*)d_in[0];
    const float* wcomp[4] = {(const float*)d_in[1], (const float*)d_in[2],
                             (const float*)d_in[3], (const float*)d_in[4]};
    const float* ucomp[4] = {(const float*)d_in[5], (const float*)d_in[6],
                             (const float*)d_in[7], (const float*)d_in[8]};
    const float* bias[4]  = {(const float*)d_in[9], (const float*)d_in[10],
                             (const float*)d_in[11], (const float*)d_in[12]};
    float* out = (float*)d_out;

    // workspace carve-up
    char*   ws   = (char*)d_ws;
    const size_t wswN = (size_t)4 * NNT * NKT * 512;           // 5,242,880 bf16
    const size_t xN   = (size_t)T_STEPS * BATCH * FEAT;        // 3,276,800 bf16
    const size_t hN   = (size_t)BATCH * HID;                   // 131,072
    __bf16* wsw  = (__bf16*)ws;
    __bf16* xbf  = (__bf16*)(ws + wswN * 2);
    __bf16* hbf0 = (__bf16*)(ws + wswN * 2 + xN * 2);
    __bf16* hbf1 = hbf0 + hN;
    float*  cst  = (float*)(ws + wswN * 2 + xN * 2 + 2 * hN * 2);

    qlstm_prep_w<<<(int)((wswN + 255) / 256), 256, 0, stream>>>(
        wcomp[0], wcomp[1], wcomp[2], wcomp[3],
        ucomp[0], ucomp[1], ucomp[2], ucomp[3], wsw);
    qlstm_prep_x<<<(int)((xN + 255) / 256), 256, 0, stream>>>(x, xbf, (int)xN);
    qlstm_zero<<<(int)((hN + 255) / 256), 256, 0, stream>>>(hbf0, cst, (int)hN);

    for (int t = 0; t < T_STEPS; ++t) {
        const __bf16* hin = (t & 1) ? hbf1 : hbf0;
        __bf16*      hout = (t & 1) ? hbf0 : hbf1;
        qlstm_step<<<NNT, 128, 0, stream>>>(xbf, wsw, hin, hout, cst,
                                            bias[0], bias[1], bias[2], bias[3],
                                            out, t);
    }
    qlstm_last<<<(int)((hN + 255) / 256), 256, 0, stream>>>(
        out + (size_t)(T_STEPS - 1) * BATCH * HID,
        out + (size_t)T_STEPS * BATCH * HID, (int)hN);
}
